// Abstract3DBUNet_57183194579692
// MI455X (gfx1250) — matrix operand order repaired and checked by reading, compiled, not run on hardware
//
#include <hip/hip_runtime.h>
#include <hip/hip_bf16.h>

typedef _Float16 f16;
typedef unsigned int u32;
typedef __attribute__((ext_vector_type(16))) _Float16 v16h;
typedef __attribute__((ext_vector_type(8)))  _Float16 v8h;
typedef __attribute__((ext_vector_type(8)))  float    v8f;
typedef __attribute__((ext_vector_type(4)))  u32      v4u;
typedef __attribute__((ext_vector_type(8)))  int      v8i;
typedef __attribute__((ext_vector_type(4)))  int      v4i;

// ---------------------------------------------------------------------------
// Layout: activations are NDHWC f16 with channel stride padded to mult of 32.
// Weights prepped to [tap(27)][Cout][CinP] f16 (cin padded with zeros).
// ---------------------------------------------------------------------------

__global__ void pack_kernel(const float* __restrict__ x, f16* __restrict__ outb, int total) {
  int i = blockIdx.x * blockDim.x + threadIdx.x;
  if (i >= total) return;
  f16* p = outb + (size_t)i * 32;
  p[0] = (f16)x[i];
  for (int c = 1; c < 32; ++c) p[c] = (f16)0.f;
}

__global__ void prep_w_kernel(const float* __restrict__ w, f16* __restrict__ outw,
                              int Cout, int Cin, int CpIn) {
  size_t i = (size_t)blockIdx.x * blockDim.x + threadIdx.x;
  const size_t total = (size_t)27 * Cout * CpIn;
  if (i >= total) return;
  const int ci = (int)(i % CpIn);
  size_t t = i / CpIn;
  const int o = (int)(t % Cout);
  const int tap = (int)(t / Cout);
  float v = 0.f;
  if (ci < Cin) v = w[((size_t)o * Cin + ci) * 27 + tap];  // OIDHW, tap=kd*9+kh*3+kw
  outw[i] = (f16)v;
}

__global__ __launch_bounds__(256)
void gn_stats_kernel(const f16* __restrict__ x, float* __restrict__ mean, float* __restrict__ rstd,
                     int Cp, int C, int G, int nvox) {
  const int b = blockIdx.x, g = blockIdx.y;
  const int Cg = C / G;
  const size_t N = (size_t)nvox * Cg;
  const f16* base = x + (size_t)b * nvox * Cp;
  float s = 0.f, ss = 0.f;
  for (size_t i = threadIdx.x; i < N; i += 256) {
    const int c = g * Cg + (int)(i % Cg);
    const size_t vox = i / Cg;
    const float v = (float)base[vox * Cp + c];
    s += v; ss += v * v;
  }
  __shared__ float rs[256], rss[256];
  rs[threadIdx.x] = s; rss[threadIdx.x] = ss; __syncthreads();
  for (int o = 128; o > 0; o >>= 1) {
    if ((int)threadIdx.x < o) { rs[threadIdx.x] += rs[threadIdx.x + o]; rss[threadIdx.x] += rss[threadIdx.x + o]; }
    __syncthreads();
  }
  if (threadIdx.x == 0) {
    const float m = rs[0] / (float)N;
    const float var = rss[0] / (float)N - m * m;
    mean[b * G + g] = m;
    rstd[b * G + g] = rsqrtf(var + 1e-5f);
  }
}

// per-(batch,channel) affine in f16 so the conv can apply GN with v_pk_fma_f16
__global__ void gn_affine_kernel(const float* __restrict__ mean, const float* __restrict__ rstd,
                                 const float* __restrict__ gamma, const float* __restrict__ beta,
                                 f16* __restrict__ scale, f16* __restrict__ shift,
                                 int C, int Cp, int G, int B) {
  int i = blockIdx.x * blockDim.x + threadIdx.x;
  if (i >= B * Cp) return;
  const int b = i / Cp, c = i % Cp;
  float sc = 0.f, sh = 0.f;
  if (c < C) {
    const int g = c / (C / G);
    sc = rstd[b * G + g] * gamma[c];
    sh = beta[c] - mean[b * G + g] * sc;
  }
  scale[i] = (f16)sc; shift[i] = (f16)sh;
}

// ---------------------------------------------------------------------------
// 1-D Tensor Data Mover copy: nelems f16 from global -> LDS (D# per ISA ch8).
// group0: count=1 | lds_addr | global_addr(57b) | type=2
// group1: data_size=1(2B); tensor_dim0=tile_dim0=nelems; 1-D (tile_dim1=0)
// groups 2/3 unused (zero) for <=2D tensors; 6-arg builtin on this toolchain.
// ---------------------------------------------------------------------------
__device__ inline void tdm_load_1d(const void* gptr, u32 lds_off, u32 nelems) {
  const unsigned long long ga = (unsigned long long)gptr;
  v4u g0;
  g0[0] = 1u;                                              // count=1
  g0[1] = lds_off;                                         // LDS byte address
  g0[2] = (u32)ga;                                         // global addr [31:0]
  g0[3] = (u32)((ga >> 32) & 0x1FFFFFFull) | (2u << 30);   // addr [56:32] | type=2
  v8i g1;
  g1[0] = (int)(1u << 16);                                 // data_size = 1 -> 2 bytes
  g1[1] = (int)((nelems & 0xFFFFu) << 16);                 // tensor_dim0[15:0]
  g1[2] = (int)(((nelems >> 16) & 0xFFFFu) | (1u << 16));  // tensor_dim0[31:16], tensor_dim1=1
  g1[3] = (int)((nelems & 0xFFFFu) << 16);                 // tile_dim0 = nelems (<=65535)
  g1[4] = 0;                                               // tile_dim1=0 (1-D), tile_dim2=0
  g1[5] = (int)nelems;                                     // tensor_dim0_stride
  g1[6] = 0; g1[7] = 0;
  const v4i z4 = {0, 0, 0, 0};
  const v8i z8 = {0, 0, 0, 0, 0, 0, 0, 0};
  __builtin_amdgcn_tensor_load_to_lds(g0, g1, z4, z4, z8, 0);
}

// ---------------------------------------------------------------------------
// Implicit-GEMM 3x3x3 conv. 4 waves/block; wave w owns M-tile blockIdx.x*4+w,
// all share NT 16-wide Cout tiles. Per tap, the B tile (16*NT x CPIN f16,
// contiguous in the prepped layout) is DMA'd to LDS by the TDM (wave 0),
// double-buffered: tensor_load_to_lds of tap t+1 overlaps WMMA of tap t,
// synchronized with s_wait_tensorcnt + workgroup barriers.
// GN is fused into the A fragment with f16 packed FMA; ReLU fused into store.
// Fragment layouts per CDNA5 ISA 7.12.2:
//   A 16x32 f16: lane m=lane&15; halves 0-7 -> K=kbA+0..7, 8-15 -> kbA+16..23.
//   B 32x16 f16: lane n=lane&15; halves 0-7 -> K=kbA+0..7, 8-15 -> kbA+16..23 (same as A).
//   C/D 16x16 f32: VGPR r -> row (lane>>4)*8+r, col lane&15.
// ---------------------------------------------------------------------------
template<int CPIN, int NT>
__global__ __launch_bounds__(128)
void conv_wmma_kernel(const f16* __restrict__ in, const f16* __restrict__ wt,
                      const f16* __restrict__ sc16, const f16* __restrict__ sh16,
                      f16* __restrict__ out, int B, int D, int H, int W, int Cout, int CpOut) {
  constexpr int TILE = 16 * NT * CPIN;      // f16 elements per B tap tile
  __shared__ f16 smem[2][TILE];
  const int lane = (int)threadIdx.x & 31;
  const int wid  = (int)threadIdx.x >> 5;
  const int mt = blockIdx.x * 4 + wid;
  const int ntBase = blockIdx.y * NT;
  const int M = B * D * H * W;
  const int hiHalf = lane >> 4;
  const int kbA = hiHalf * 8;
  const int kbB = hiHalf * 16;
  int s = mt * 16 + (lane & 15);
  if (s >= M) s = M - 1;                    // clamp ragged tail (stores guarded)
  int w = s % W; int t = s / W;
  int h = t % H; t /= H;
  int d = t % D; const int b = t / D;
  const f16* scb = sc16 + b * CPIN;
  const f16* shb = sh16 + b * CPIN;
  const int n = lane & 15;
  const f16* wbase = wt + (size_t)(ntBase * 16) * CPIN;   // + tap*Cout*CPIN per tap
  v8f acc[NT];
#pragma unroll
  for (int j = 0; j < NT; ++j) {
    const v8f z = {0.f, 0.f, 0.f, 0.f, 0.f, 0.f, 0.f, 0.f};
    acc[j] = z;
  }
  if (wid == 0)
    tdm_load_1d(wbase, (u32)(uintptr_t)&smem[0][0], TILE);
  for (int tap = 0; tap < 27; ++tap) {
    const int cur = tap & 1;
    if (wid == 0) __builtin_amdgcn_s_wait_tensorcnt(0);   // buffer `cur` landed
    __syncthreads();
    if (wid == 0 && tap + 1 < 27)                         // prefetch next tap's B tile
      tdm_load_1d(wbase + (size_t)(tap + 1) * Cout * CPIN,
                  (u32)(uintptr_t)&smem[cur ^ 1][0], TILE);
    const int kd = tap / 9 - 1, kh = (tap / 3) % 3 - 1, kw = tap % 3 - 1;
    const int dd = d + kd, hh = h + kh, ww = w + kw;
    const bool inb = (unsigned)dd < (unsigned)D && (unsigned)hh < (unsigned)H && (unsigned)ww < (unsigned)W;
    const f16* src = in + (size_t)(((b * D + dd) * H + hh) * W + ww) * CPIN;
#pragma unroll
    for (int k0 = 0; k0 < CPIN; k0 += 32) {
      v16h afrag;
      if (inb) {
        const v8h lo = *(const v8h*)(src + k0 + kbA);        // 16B contiguous
        const v8h hi = *(const v8h*)(src + k0 + kbA + 16);
        const v8h scl = *(const v8h*)(scb + k0 + kbA);       // tiny, cache-resident
        const v8h sch = *(const v8h*)(scb + k0 + kbA + 16);
        const v8h shl = *(const v8h*)(shb + k0 + kbA);
        const v8h shh = *(const v8h*)(shb + k0 + kbA + 16);
        const v8h alo = lo * scl + shl;                      // v_pk_fma_f16 (fused GN)
        const v8h ahi = hi * sch + shh;
        afrag = __builtin_shufflevector(alo, ahi, 0,1,2,3,4,5,6,7,8,9,10,11,12,13,14,15);
      } else {
#pragma unroll
        for (int i = 0; i < 16; ++i) afrag[i] = (f16)0.f;    // conv zero-pad
      }
#pragma unroll
      for (int j = 0; j < NT; ++j) {
        const f16* bsrc = &smem[cur][(j * 16 + n) * CPIN + k0 + kbA];
        const v8h blo = *(const v8h*)bsrc;                   // ds_load_b128
        const v8h bhi = *(const v8h*)(bsrc + 16);
        const v16h bfrag = __builtin_shufflevector(blo, bhi, 0,1,2,3,4,5,6,7,8,9,10,11,12,13,14,15);
        acc[j] = __builtin_amdgcn_wmma_f32_16x16x32_f16(false, afrag, false, bfrag,
                                                        (short)0, acc[j], false, false);
      }
    }
    __syncthreads();   // all waves done reading `cur` before it is refilled
  }
#pragma unroll
  for (int j = 0; j < NT; ++j) {
#pragma unroll
    for (int r = 0; r < 8; ++r) {
      const int mr = hiHalf * 8 + r;
      const int so = mt * 16 + mr;
      if (so < M) {
        float v = acc[j][r];
        v = v > 0.f ? v : 0.f;   // fused ReLU
        out[(size_t)so * CpOut + (ntBase + j) * 16 + n] = (f16)v;
      }
    }
  }
}

__global__ void maxpool_kernel(const f16* __restrict__ in, f16* __restrict__ outb,
                               int B, int D, int H, int W, int C) {
  const int Do = D >> 1, Ho = H >> 1, Wo = W >> 1;
  const size_t total = (size_t)B * Do * Ho * Wo * C;
  size_t i = (size_t)blockIdx.x * blockDim.x + threadIdx.x;
  if (i >= total) return;
  const int c = (int)(i % C); size_t t = i / C;
  const int w = (int)(t % Wo); t /= Wo;
  const int h = (int)(t % Ho); t /= Ho;
  const int d = (int)(t % Do); const int b = (int)(t / Do);
  float m = -3.4e38f;
  for (int dz = 0; dz < 2; ++dz)
    for (int dy = 0; dy < 2; ++dy)
      for (int dx = 0; dx < 2; ++dx) {
        const float v = (float)in[(size_t)((((b * D + 2 * d + dz) * H + 2 * h + dy) * W + 2 * w + dx)) * C + c];
        if (v > m) m = v;
      }
  outb[i] = (f16)m;
}

__global__ void upcat_kernel(const f16* __restrict__ skip, const f16* __restrict__ low,
                             f16* __restrict__ outb, int B, int D, int H, int W, int Cs, int Cl) {
  const int Ct = Cs + Cl;
  const size_t total = (size_t)B * D * H * W * Ct;
  size_t i = (size_t)blockIdx.x * blockDim.x + threadIdx.x;
  if (i >= total) return;
  const int c = (int)(i % Ct); size_t t = i / Ct;
  const int w = (int)(t % W); t /= W;
  const int h = (int)(t % H); t /= H;
  const int d = (int)(t % D); const int b = (int)(t / D);
  f16 v;
  if (c < Cs) v = skip[(size_t)((((b * D + d) * H + h) * W + w)) * Cs + c];
  else        v = low [(size_t)((((b * (D >> 1) + (d >> 1)) * (H >> 1) + (h >> 1)) * (W >> 1) + (w >> 1))) * Cl + (c - Cs)];
  outb[i] = v;
}

// flat index j of (B,256,9,9,6) NCDHW <-> our NDHWC buffer (stride 256)
__global__ __launch_bounds__(256)
void latent_proj_kernel(const f16* __restrict__ act, const float* __restrict__ wt,
                        const float* __restrict__ bias, float* __restrict__ outv, int R) {
  const int b = blockIdx.x / R;
  const int r = blockIdx.x % R;
  const float* wrow = wt + (size_t)r * 124416;
  float s = 0.f;
  for (int j = threadIdx.x; j < 124416; j += 256) {
    const int c = j / 486;
    const int rem = j % 486;
    const int dd = rem / 54, rem2 = rem % 54;
    const int hh = rem2 / 6, ww = rem2 % 6;
    const float v = (float)act[(size_t)(((b * 9 + dd) * 9 + hh) * 6 + ww) * 256 + c];
    s += v * wrow[j];
  }
  __shared__ float red[256];
  red[threadIdx.x] = s; __syncthreads();
  for (int o = 128; o > 0; o >>= 1) {
    if ((int)threadIdx.x < o) red[threadIdx.x] += red[threadIdx.x + o];
    __syncthreads();
  }
  if (threadIdx.x == 0) outv[b * R + r] = red[0] + bias[r];
}

__device__ inline void jacobi32(float* A, float* V, int t) {
  for (int i = 0; i < 32; ++i) V[i * 32 + t] = (i == t) ? 1.f : 0.f;
  __syncthreads();
  for (int sweep = 0; sweep < 8; ++sweep) {
    for (int p = 0; p < 31; ++p) {
      for (int q = p + 1; q < 32; ++q) {
        const float apq = A[p * 32 + q];
        const float app = A[p * 32 + p];
        const float aqq = A[q * 32 + q];
        __syncthreads();
        if (fabsf(apq) > 1e-12f) {  // uniform branch (LDS value)
          const float theta = 0.5f * (aqq - app) / apq;
          const float tt = ((theta >= 0.f) ? 1.f : -1.f) / (fabsf(theta) + sqrtf(theta * theta + 1.f));
          const float c = rsqrtf(tt * tt + 1.f);
          const float sn = tt * c;
          const float aip = A[t * 32 + p], aiq = A[t * 32 + q];
          __syncthreads();
          A[t * 32 + p] = c * aip - sn * aiq;
          A[t * 32 + q] = sn * aip + c * aiq;
          __syncthreads();
          const float api = A[p * 32 + t], aqi = A[q * 32 + t];
          __syncthreads();
          A[p * 32 + t] = c * api - sn * aqi;
          A[q * 32 + t] = sn * api + c * aqi;
          const float vip = V[t * 32 + p], viq = V[t * 32 + q];
          V[t * 32 + p] = c * vip - sn * viq;
          V[t * 32 + q] = sn * vip + c * viq;
          __syncthreads();
        }
      }
    }
  }
  __syncthreads();
}

__global__ __launch_bounds__(32)
void fixcov_kernel(const float* __restrict__ lv, const float* __restrict__ mu,
                   const float* __restrict__ eps, float* __restrict__ sample) {
  __shared__ float A[1024], V[1024], C2[1024], wn[32];
  __shared__ float sneg, pmin;
  const int b = blockIdx.x, t = threadIdx.x;
  for (int i = 0; i < 32; ++i) {  // C = exp(lv/2), sym_lower
    const float lo = __expf(0.5f * lv[b * 1024 + i * 32 + t]);
    const float hi = __expf(0.5f * lv[b * 1024 + t * 32 + i]);
    A[i * 32 + t] = (i >= t) ? lo : hi;
  }
  __syncthreads();
  jacobi32(A, V, t);
  for (int i = 0; i < 32; ++i) {  // C1 = V diag(w) V^T (inv(V)=V^T, orthogonal)
    float s = 0.f;
    for (int k = 0; k < 32; ++k) s += V[i * 32 + k] * A[k * 32 + k] * V[t * 32 + k];
    C2[i * 32 + t] = s;
  }
  __syncthreads();
  for (int i = 0; i < 32; ++i) A[i * 32 + t] = (i >= t) ? C2[i * 32 + t] : C2[t * 32 + i];
  __syncthreads();
  jacobi32(A, V, t);
  const float w2 = A[t * 32 + t];
  wn[t] = w2;
  __syncthreads();
  if (t == 0) {
    float s = 0.f;
    for (int k = 0; k < 32; ++k) s += (wn[k] < 0.f) ? 1.f : 0.f;
    sneg = s;
    float p = __builtin_inff();
    for (int k = 0; k < 32; ++k) if (wn[k] > 0.f && wn[k] < p) p = wn[k];
    pmin = (p == __builtin_inff()) ? 0.f : p;
  }
  __syncthreads();
  {
    const float tden = sneg * sneg * 100.f + 1.f;
    const float nw = (w2 < 0.f) ? pmin * (sneg - w2) * (sneg - w2) / tden : w2;
    __syncthreads();
    wn[t] = nw;
  }
  __syncthreads();
  for (int i = 0; i < 32; ++i) {  // C2 = V2 diag(w_new) V2^T
    float s = 0.f;
    for (int k = 0; k < 32; ++k) s += V[i * 32 + k] * wn[k] * V[t * 32 + k];
    C2[i * 32 + t] = s;
  }
  __syncthreads();
  for (int k = 0; k < 32; ++k) {  // in-place lower Cholesky
    if (t == 0) C2[k * 32 + k] = sqrtf(fmaxf(C2[k * 32 + k], 1e-20f));
    __syncthreads();
    if (t > k) C2[t * 32 + k] /= C2[k * 32 + k];
    __syncthreads();
    for (int j = k + 1; j < 32; ++j)
      if (t >= j) C2[t * 32 + j] -= C2[t * 32 + k] * C2[j * 32 + k];
    __syncthreads();
  }
  float s = mu[b * 32 + t];
  for (int j = 0; j <= t; ++j) s += C2[t * 32 + j] * eps[b * 32 + j];
  sample[b * 32 + t] = s;
}

__global__ void latent_decode_kernel(const float* __restrict__ sample, const float* __restrict__ wt,
                                     const float* __restrict__ bias, f16* __restrict__ outb) {
  const int i = blockIdx.x * blockDim.x + threadIdx.x;
  if (i >= 2 * 124416) return;
  const int b = i / 124416, j = i % 124416;
  const float* wr = wt + (size_t)j * 32;
  const float* sp = sample + b * 32;
  float s = bias[j];
  for (int k = 0; k < 32; ++k) s += sp[k] * wr[k];
  const int c = j / 486, rem = j % 486;
  const int dd = rem / 54, rem2 = rem % 54;
  const int hh = rem2 / 6, ww = rem2 % 6;
  outb[(size_t)(((b * 9 + dd) * 9 + hh) * 6 + ww) * 256 + c] = (f16)s;
}

__global__ void final_kernel(const f16* __restrict__ in, const float* __restrict__ fw,
                             const float* __restrict__ fb, float* __restrict__ outv, int total) {
  int i = blockIdx.x * blockDim.x + threadIdx.x;
  if (i >= total) return;
  float s = fb[0];
  const f16* p = in + (size_t)i * 32;
  for (int c = 0; c < 32; ++c) s += (float)p[c] * fw[c];
  outv[i] = 1.f / (1.f + __expf(-s));
}

// ---------------------------------------------------------------------------
extern "C" void kernel_launch(void* const* d_in, const int* in_sizes, int n_in,
                              void* d_out, int out_size, void* d_ws, size_t ws_size,
                              hipStream_t stream) {
  (void)in_sizes; (void)n_in; (void)out_size; (void)ws_size;
  int ix = 0;
  const float* x    = (const float*)d_in[ix++];
  const float* epsL = (const float*)d_in[ix++];
  struct DCp { const float *g1, *b1, *w1, *g2, *b2, *w2; };
  DCp enc[4], dec[3];
  for (int i = 0; i < 4; ++i) {
    enc[i].g1 = (const float*)d_in[ix++]; enc[i].b1 = (const float*)d_in[ix++];
    enc[i].w1 = (const float*)d_in[ix++]; enc[i].g2 = (const float*)d_in[ix++];
    enc[i].b2 = (const float*)d_in[ix++]; enc[i].w2 = (const float*)d_in[ix++];
  }
  for (int i = 0; i < 3; ++i) {
    dec[i].g1 = (const float*)d_in[ix++]; dec[i].b1 = (const float*)d_in[ix++];
    dec[i].w1 = (const float*)d_in[ix++]; dec[i].g2 = (const float*)d_in[ix++];
    dec[i].b2 = (const float*)d_in[ix++]; dec[i].w2 = (const float*)d_in[ix++];
  }
  const float* mu_w  = (const float*)d_in[ix++];
  const float* mu_b  = (const float*)d_in[ix++];
  const float* lv_w  = (const float*)d_in[ix++];
  const float* lv_b  = (const float*)d_in[ix++];
  const float* l2d_w = (const float*)d_in[ix++];
  const float* l2d_b = (const float*)d_in[ix++];
  const float* fin_w = (const float*)d_in[ix++];
  const float* fin_b = (const float*)d_in[ix++];

  char* ws = (char*)d_ws; size_t off = 0;
  auto alloc = [&](size_t nbytes) -> char* {
    char* p = ws + off; off += (nbytes + 255) & ~(size_t)255; return p;
  };
  const int pos0 = 497664, pos1 = 62208, pos2 = 7776, pos3 = 972;
  f16* bA    = (f16*)alloc((size_t)pos0 * 32 * 2);    // packed input, later d0a
  f16* bB    = (f16*)alloc((size_t)pos0 * 32 * 2);    // enc0a, later d0b
  f16* bC    = (f16*)alloc((size_t)pos0 * 32 * 2);    // enc0b (skip L0)
  f16* bCat0 = (f16*)alloc((size_t)pos0 * 96 * 2);
  f16* bP1   = (f16*)alloc((size_t)pos1 * 32 * 2);
  f16* bE1a  = (f16*)alloc((size_t)pos1 * 32 * 2);
  f16* bE1b  = (f16*)alloc((size_t)pos1 * 64 * 2);    // skip L1
  f16* bCat1 = (f16*)alloc((size_t)pos1 * 192 * 2);
  f16* bD1a  = (f16*)alloc((size_t)pos1 * 64 * 2);
  f16* bD1b  = (f16*)alloc((size_t)pos1 * 64 * 2);
  f16* bP2   = (f16*)alloc((size_t)pos2 * 64 * 2);
  f16* bE2a  = (f16*)alloc((size_t)pos2 * 64 * 2);
  f16* bE2b  = (f16*)alloc((size_t)pos2 * 128 * 2);   // skip L2
  f16* bCat2 = (f16*)alloc((size_t)pos2 * 384 * 2);
  f16* bD2a  = (f16*)alloc((size_t)pos2 * 128 * 2);
  f16* bD2b  = (f16*)alloc((size_t)pos2 * 128 * 2);
  f16* bP3   = (f16*)alloc((size_t)pos3 * 128 * 2);
  f16* bE3a  = (f16*)alloc((size_t)pos3 * 128 * 2);
  f16* bE3b  = (f16*)alloc((size_t)pos3 * 256 * 2);
  f16* bDL   = (f16*)alloc((size_t)pos3 * 256 * 2);
  f16* bW    = (f16*)alloc((size_t)27 * 384 * 128 * 2);  // largest prepped weight
  float* meanB = (float*)alloc(16 * 4);
  float* rstdB = (float*)alloc(16 * 4);
  f16*   scB   = (f16*)alloc(2 * 384 * 2);
  f16*   shB   = (f16*)alloc(2 * 384 * 2);
  float* muB   = (float*)alloc(64 * 4);
  float* lvB   = (float*)alloc(2048 * 4);
  float* smB   = (float*)alloc(64 * 4);

  auto conv = [&](const f16* in, const float* wgt, const float* gm, const float* bt,
                  f16* outb, int D, int H, int W, int Cin, int CpIn, int Cout, int CpOut) {
    const size_t wtot = (size_t)27 * Cout * CpIn;
    prep_w_kernel<<<dim3((unsigned)((wtot + 255) / 256)), dim3(256), 0, stream>>>(wgt, bW, Cout, Cin, CpIn);
    const int G = (Cin % 8 == 0) ? 8 : 1;
    gn_stats_kernel<<<dim3(2, G), dim3(256), 0, stream>>>(in, meanB, rstdB, CpIn, Cin, G, D * H * W);
    const int aftot = 2 * CpIn;
    gn_affine_kernel<<<dim3((aftot + 63) / 64), dim3(64), 0, stream>>>(meanB, rstdB, gm, bt, scB, shB, Cin, CpIn, G, 2);
    const int M = 2 * D * H * W;
    const int NT = (Cout % 32 == 0) ? 2 : 1;
    dim3 grid((unsigned)((M + 63) / 64), (unsigned)(Cout / (16 * NT)));
    dim3 blk(128);
    if (NT == 1) {
      if (CpIn == 32) conv_wmma_kernel<32, 1><<<grid, blk, 0, stream>>>(in, bW, scB, shB, outb, 2, D, H, W, Cout, CpOut);
    } else {
      switch (CpIn) {
        case 32:  conv_wmma_kernel<32,  2><<<grid, blk, 0, stream>>>(in, bW, scB, shB, outb, 2, D, H, W, Cout, CpOut); break;
        case 64:  conv_wmma_kernel<64,  2><<<grid, blk, 0, stream>>>(in, bW, scB, shB, outb, 2, D, H, W, Cout, CpOut); break;
        case 96:  conv_wmma_kernel<96,  2><<<grid, blk, 0, stream>>>(in, bW, scB, shB, outb, 2, D, H, W, Cout, CpOut); break;
        case 128: conv_wmma_kernel<128, 2><<<grid, blk, 0, stream>>>(in, bW, scB, shB, outb, 2, D, H, W, Cout, CpOut); break;
        case 192: conv_wmma_kernel<192, 2><<<grid, blk, 0, stream>>>(in, bW, scB, shB, outb, 2, D, H, W, Cout, CpOut); break;
        case 384: conv_wmma_kernel<384, 2><<<grid, blk, 0, stream>>>(in, bW, scB, shB, outb, 2, D, H, W, Cout, CpOut); break;
        default: break;
      }
    }
  };
  auto ew = [](size_t total) { return dim3((unsigned)((total + 255) / 256)); };

  // ---- encoder ----
  pack_kernel<<<ew(pos0), dim3(256), 0, stream>>>(x, bA, pos0);
  conv(bA,  enc[0].w1, enc[0].g1, enc[0].b1, bB,   72, 72, 48, 1,   32,  16,  32);
  conv(bB,  enc[0].w2, enc[0].g2, enc[0].b2, bC,   72, 72, 48, 16,  32,  32,  32);
  maxpool_kernel<<<ew((size_t)pos1 * 32), dim3(256), 0, stream>>>(bC, bP1, 2, 72, 72, 48, 32);
  conv(bP1, enc[1].w1, enc[1].g1, enc[1].b1, bE1a, 36, 36, 24, 32,  32,  32,  32);
  conv(bE1a,enc[1].w2, enc[1].g2, enc[1].b2, bE1b, 36, 36, 24, 32,  32,  64,  64);
  maxpool_kernel<<<ew((size_t)pos2 * 64), dim3(256), 0, stream>>>(bE1b, bP2, 2, 36, 36, 24, 64);
  conv(bP2, enc[2].w1, enc[2].g1, enc[2].b1, bE2a, 18, 18, 12, 64,  64,  64,  64);
  conv(bE2a,enc[2].w2, enc[2].g2, enc[2].b2, bE2b, 18, 18, 12, 64,  64,  128, 128);
  maxpool_kernel<<<ew((size_t)pos3 * 128), dim3(256), 0, stream>>>(bE2b, bP3, 2, 18, 18, 12, 128);
  conv(bP3, enc[3].w1, enc[3].g1, enc[3].b1, bE3a, 9, 9, 6,    128, 128, 128, 128);
  conv(bE3a,enc[3].w2, enc[3].g2, enc[3].b2, bE3b, 9, 9, 6,    128, 128, 256, 256);

  // ---- latent ----
  latent_proj_kernel<<<dim3(2 * 32),   dim3(256), 0, stream>>>(bE3b, mu_w, mu_b, muB, 32);
  latent_proj_kernel<<<dim3(2 * 1024), dim3(256), 0, stream>>>(bE3b, lv_w, lv_b, lvB, 1024);
  fixcov_kernel<<<dim3(2), dim3(32), 0, stream>>>(lvB, muB, epsL, smB);
  latent_decode_kernel<<<ew((size_t)2 * 124416), dim3(256), 0, stream>>>(smB, l2d_w, l2d_b, bDL);

  // ---- decoder ----
  upcat_kernel<<<ew((size_t)pos2 * 384), dim3(256), 0, stream>>>(bE2b, bDL,  bCat2, 2, 18, 18, 12, 128, 256);
  conv(bCat2, dec[0].w1, dec[0].g1, dec[0].b1, bD2a, 18, 18, 12, 384, 384, 128, 128);
  conv(bD2a,  dec[0].w2, dec[0].g2, dec[0].b2, bD2b, 18, 18, 12, 128, 128, 128, 128);
  upcat_kernel<<<ew((size_t)pos1 * 192), dim3(256), 0, stream>>>(bE1b, bD2b, bCat1, 2, 36, 36, 24, 64, 128);
  conv(bCat1, dec[1].w1, dec[1].g1, dec[1].b1, bD1a, 36, 36, 24, 192, 192, 64, 64);
  conv(bD1a,  dec[1].w2, dec[1].g2, dec[1].b2, bD1b, 36, 36, 24, 64,  64,  64, 64);
  upcat_kernel<<<ew((size_t)pos0 * 96), dim3(256), 0, stream>>>(bC, bD1b, bCat0, 2, 72, 72, 48, 32, 64);
  conv(bCat0, dec[2].w1, dec[2].g1, dec[2].b1, bA,   72, 72, 48, 96,  96,  32, 32);
  conv(bA,    dec[2].w2, dec[2].g2, dec[2].b2, bB,   72, 72, 48, 32,  32,  32, 32);

  // ---- final 1x1x1 conv + sigmoid (f32 out, NCDHW with C=1 == NDHW) ----
  final_kernel<<<ew(pos0), dim3(256), 0, stream>>>(bB, fin_w, fin_b, (float*)d_out, pos0);
}
